// MemoryModule_36051955482705
// MI455X (gfx1250) — matrix operand order repaired and checked by reading, not compiled, not run on hardware
//
#include <hip/hip_runtime.h>
#include <hip/hip_bf16.h>

typedef __bf16 bf16_t;
typedef __attribute__((ext_vector_type(16))) __bf16 v16bf;
typedef __attribute__((ext_vector_type(8)))  __bf16 v8bf;
typedef __attribute__((ext_vector_type(8)))  float  v8f;
typedef __attribute__((ext_vector_type(2)))  float  v2f;

#define B_ROWS   16384
#define IN_DIM   512
#define S_MEM    32
#define D_DIM    128
#define TILE_M   16

// workspace byte offsets
#define W1FRAG_OFF 0          // bf16 [8 ntile][16 kstep][32 lane][16 e]  = 128KB
#define W2FRAG_OFF 131072     // bf16 [8 ntile][4 kstep][32 lane][16 e]   = 32KB
#define MPART_OFF  163840     // f32  [32][128]                           = 16KB
#define ACC_OFF    180224     // f32  gate[4096] ++ write[4096]           = 32KB

__device__ __forceinline__ float wave_reduce_add(float v) {
  for (int off = 16; off > 0; off >>= 1) v += __shfl_xor(v, off, 32);
  return v;
}
__device__ __forceinline__ float wave_reduce_max(float v) {
  for (int off = 16; off > 0; off >>= 1) v = fmaxf(v, __shfl_xor(v, off, 32));
  return v;
}

// ---------------- prep: weight swizzle into WMMA B-fragment order + mem_part ----------------
__global__ void prep_kernel(const float* __restrict__ w_in, const float* __restrict__ w_gate,
                            const float* __restrict__ mem, const float* __restrict__ b_gate,
                            bf16_t* __restrict__ w1f, bf16_t* __restrict__ w2f,
                            float* __restrict__ mpart, float* __restrict__ acc) {
  int tid = blockIdx.x * blockDim.x + threadIdx.x;   // 65536 threads
  {
    // w_in [512][128] -> B frag: lane l holds col n = 16*ntile + l%16,
    // K = 32*kstep + 8*(l/16) + e for e<8, 32*kstep + 16 + 8*(l/16) + (e-8) for e>=8
    int e = tid & 15, lane = (tid >> 4) & 31, kstep = (tid >> 9) & 15, ntile = tid >> 13;
    int n = ntile * 16 + (lane & 15);
    int k = kstep * 32 + 8 * (lane >> 4) + (e & 7) + 16 * (e >> 3);
    w1f[tid] = (bf16_t)w_in[k * D_DIM + n];
  }
  if (tid < 16384) {
    int e = tid & 15, lane = (tid >> 4) & 31, kstep = (tid >> 9) & 3, ntile = tid >> 11;
    int n = ntile * 16 + (lane & 15);
    int k = kstep * 32 + 8 * (lane >> 4) + (e & 7) + 16 * (e >> 3);
    w2f[tid] = (bf16_t)w_gate[(D_DIM + k) * D_DIM + n];   // rows D..2D-1 multiply x_proj
  }
  if (tid < 8192) acc[tid] = 0.0f;                         // zero gate/write accumulators
  if (tid < 4096) {                                        // mem_part = mem @ w_gate[:D] + b_gate
    int s = tid >> 7, d = tid & 127;
    float a = b_gate[d];
    for (int k = 0; k < D_DIM; ++k) a += mem[s * D_DIM + k] * w_gate[k * D_DIM + d];
    mpart[tid] = a;
  }
}

// ---------------- main fused kernel: 16 rows of B per block, 8 waves ----------------
__global__ __launch_bounds__(256) void main_kernel(
    const float* __restrict__ x, const float* __restrict__ mem,
    const float* __restrict__ ln_g, const float* __restrict__ ln_b,
    const float* __restrict__ b_in,
    const bf16_t* __restrict__ w1f, const bf16_t* __restrict__ w2f,
    const float* __restrict__ mpart,
    float* __restrict__ gate_sum, float* __restrict__ write_sum,
    float* __restrict__ out) {
  __shared__ bf16_t a_bf[TILE_M][IN_DIM];     // 16KB  LN(x) in A-fragment-friendly row-major
  __shared__ float  memLds[S_MEM][D_DIM];     // 16KB
  __shared__ float  xp[TILE_M][D_DIM];        // 8KB   x_proj fp32
  __shared__ bf16_t xp_bf[TILE_M][D_DIM];     // 4KB
  __shared__ float  probs[TILE_M][S_MEM];     // 2KB
  __shared__ float  xpart[TILE_M][D_DIM];     // 8KB

  const int t = threadIdx.x;
  const int wave = t >> 5, lane = t & 31;
  const int half = lane >> 4, l16 = lane & 15;
  const int row0 = blockIdx.x * TILE_M;

  for (int i = t; i < S_MEM * D_DIM; i += 256) ((float*)memLds)[i] = mem[i];

  // ---- LayerNorm, one pass, values kept in registers; wave owns rows 2w, 2w+1 ----
  for (int rr = 0; rr < 2; ++rr) {
    const int r = 2 * wave + rr;
    const float* xr = x + (size_t)(row0 + r) * IN_DIM;
    float va[16];
    float s = 0.f, s2 = 0.f;
    for (int i = 0; i < 16; ++i) {
      float v = xr[lane + 32 * i];
      va[i] = v; s += v; s2 += v * v;
    }
    s = wave_reduce_add(s); s2 = wave_reduce_add(s2);
    float mu   = s  * (1.0f / IN_DIM);
    float var  = s2 * (1.0f / IN_DIM) - mu * mu;
    float rstd = rsqrtf(var + 1e-5f);
    for (int i = 0; i < 16; ++i) {
      int k = lane + 32 * i;
      a_bf[r][k] = (bf16_t)((va[i] - mu) * rstd * ln_g[k] + ln_b[k]);
    }
  }
  __syncthreads();

  // ---- GEMM1: x_proj = LN(x)[16x512] @ w_in[512x128] + b_in  (bf16 WMMA, K loop of 16) ----
  {
    v8f c = {};
    const int n0 = wave * 16;
    for (int ks = 0; ks < 16; ++ks) {
      const int k0 = ks * 32;
      union { v16bf v; v8bf h[2]; } A;   // 16-bit A 16x32: two contiguous 8-elem chunks/lane
      A.h[0] = *(const v8bf*)&a_bf[l16][k0 + 8 * half];
      A.h[1] = *(const v8bf*)&a_bf[l16][k0 + 16 + 8 * half];
      v16bf Bv = *(const v16bf*)(w1f + (((size_t)wave * 16 + ks) * 32 + lane) * 16);
      c = __builtin_amdgcn_wmma_f32_16x16x32_bf16(false, A.v, false, Bv, (short)0, c, false, false);
    }
    for (int r = 0; r < 8; ++r) {        // C layout: M = r + 8*half, N = n0 + l16
      int m = r + 8 * half, n = n0 + l16;
      float v = c[r] + b_in[n];
      xp[m][n] = v;
      xp_bf[m][n] = (bf16_t)v;
    }
  }
  __syncthreads();

  // ---- scores = x_proj @ mem^T, then softmax over S=32 ----
  for (int p = t; p < TILE_M * S_MEM; p += 256) {
    int m = p >> 5, s = p & 31;
    float acc = 0.f;
    for (int d = 0; d < D_DIM; ++d) acc += xp[m][d] * memLds[s][d];
    probs[m][s] = acc;
  }
  __syncthreads();
  for (int rr = 0; rr < 2; ++rr) {
    int m = 2 * wave + rr;
    float sc = probs[m][lane];
    float mx = wave_reduce_max(sc);
    float e  = __expf(sc - mx);
    float se = wave_reduce_add(e);
    probs[m][lane] = e / se;
  }
  __syncthreads();

  // ---- read_content = probs[16x32] @ mem[32x128]  (exact fp32 WMMA 16x16x4, 8 K-steps) ----
  {
    v8f c = {};
    const int n0 = wave * 16;
    for (int ks = 0; ks < 8; ++ks) {
      int ka = ks * 4 + 2 * half;        // f32 A 16x4 / B 4x16: K = 2*half + j
      v2f A, Bv;
      A.x  = probs[l16][ka];       A.y  = probs[l16][ka + 1];
      Bv.x = memLds[ka][n0 + l16]; Bv.y = memLds[ka + 1][n0 + l16];
      c = __builtin_amdgcn_wmma_f32_16x16x4_f32(false, A, false, Bv, (short)0, c, false, false);
    }
    for (int r = 0; r < 8; ++r) {
      int m = r + 8 * half;
      out[(size_t)(row0 + m) * D_DIM + n0 + l16] = c[r];
    }
  }

  // ---- xp_part = x_proj[16x128] @ w_gate[D:,:][128x128]  (bf16 WMMA, 4 K-steps) ----
  {
    v8f c = {};
    const int n0 = wave * 16;
    for (int ks = 0; ks < 4; ++ks) {
      const int k0 = ks * 32;
      union { v16bf v; v8bf h[2]; } A;
      A.h[0] = *(const v8bf*)&xp_bf[l16][k0 + 8 * half];
      A.h[1] = *(const v8bf*)&xp_bf[l16][k0 + 16 + 8 * half];
      v16bf Bv = *(const v16bf*)(w2f + (((size_t)wave * 4 + ks) * 32 + lane) * 16);
      c = __builtin_amdgcn_wmma_f32_16x16x32_bf16(false, A.v, false, Bv, (short)0, c, false, false);
    }
    for (int r = 0; r < 8; ++r) xpart[r + 8 * half][wave * 16 + l16] = c[r];
  }
  __syncthreads();

  // ---- block-local reduction over 16 rows: sigmoid gate + write-info, then atomics ----
  for (int i = t; i < S_MEM * D_DIM; i += 256) {
    int s = i >> 7, d = i & 127;
    float mp = mpart[i];
    float g = 0.f, wsum = 0.f;
    for (int m = 0; m < TILE_M; ++m) {
      float z = mp + xpart[m][d];
      g    += 1.0f / (1.0f + __expf(-z));
      wsum += probs[m][s] * xp[m][d];
    }
    atomicAdd(&gate_sum[i], g);
    atomicAdd(&write_sum[i], wsum);
  }
}

// ---------------- finish: new_memory blend ----------------
__global__ void finish_kernel(const float* __restrict__ mem, const float* __restrict__ acc,
                              float* __restrict__ out) {
  int i = blockIdx.x * blockDim.x + threadIdx.x;
  if (i < S_MEM * D_DIM) {
    float g = acc[i]        * (1.0f / B_ROWS);
    float w = acc[4096 + i] * (1.0f / B_ROWS);
    out[(size_t)B_ROWS * D_DIM + i] = mem[i] * (1.0f - g) + w * g;
  }
}

extern "C" void kernel_launch(void* const* d_in, const int* in_sizes, int n_in,
                              void* d_out, int out_size, void* d_ws, size_t ws_size,
                              hipStream_t stream) {
  const float* x      = (const float*)d_in[0];
  const float* mem    = (const float*)d_in[1];
  const float* ln_g   = (const float*)d_in[2];
  const float* ln_b   = (const float*)d_in[3];
  const float* w_in   = (const float*)d_in[4];
  const float* b_in   = (const float*)d_in[5];
  const float* w_gate = (const float*)d_in[6];
  const float* b_gate = (const float*)d_in[7];
  float* out = (float*)d_out;
  char* ws = (char*)d_ws;
  bf16_t* w1f   = (bf16_t*)(ws + W1FRAG_OFF);
  bf16_t* w2f   = (bf16_t*)(ws + W2FRAG_OFF);
  float*  mpart = (float*)(ws + MPART_OFF);
  float*  acc   = (float*)(ws + ACC_OFF);

  prep_kernel<<<256, 256, 0, stream>>>(w_in, w_gate, mem, b_gate, w1f, w2f, mpart, acc);
  main_kernel<<<B_ROWS / TILE_M, 256, 0, stream>>>(x, mem, ln_g, ln_b, b_in, w1f, w2f,
                                                   mpart, acc, acc + 4096, out);
  finish_kernel<<<16, 256, 0, stream>>>(mem, acc, out);
}
